// PatchInpainting_12876311953622
// MI455X (gfx1250) — matrix operand order repaired and checked by reading, not compiled, not run on hardware
//
#include <hip/hip_runtime.h>
#include <hip/hip_bf16.h>

// ---------------------------------------------------------------------------
// MI455X (gfx1250) top-k attention, compile-only optimized design.
//   HBM floor: 0.54GB attn_probs write ~= 23us @ 23.3TB/s  -> write probs once.
//   Compute:   ~45 GFLOP -> f16 WMMA (v_wmma_f32_16x16x32_f16), f32 accum.
//   Logits tile (16 x 2048 f32 = 128KB) lives in LDS (320KB/WGP) - never HBM.
//   probs @ V is 32/2048 sparse after top-k -> scalar sparse gather.
//   K-loops are register double-buffered so loads overlap WMMAs (partial
//   s_wait_loadcnt instead of drain-to-zero before every matrix op).
// ---------------------------------------------------------------------------

typedef __attribute__((ext_vector_type(8)))  _Float16 v8h;
typedef __attribute__((ext_vector_type(16))) _Float16 v16h;
typedef __attribute__((ext_vector_type(8)))  float    v8f;

// ---- WMMA fragment loaders (ISA 7.12.2 lane layouts, wave32) --------------
// A 16x32 f16: lane L<16 -> M=L, K={k0..k0+7, k0+16..k0+23}; lanes 16..31 +8.
__device__ __forceinline__ v16h load_a_frag(const _Float16* rowBase, int sel, int k0) {
  const _Float16* p = rowBase + k0 + sel * 8;
  v8h lo = *(const v8h*)(p);        // 16B global_load_b128
  v8h hi = *(const v8h*)(p + 16);
  v16h r;
#pragma unroll
  for (int i = 0; i < 8; ++i) { r[i] = lo[i]; r[i + 8] = hi[i]; }
  return r;
}
// B 32x16 f16 (B stored as rows of the N-matrix, K contiguous):
// lane L<16 -> N=L, K={k0..k0+7, k0+16..k0+23}; lanes 16..31 +8.
__device__ __forceinline__ v16h load_b_frag(const _Float16* rowBase, int sel, int k0) {
  const _Float16* p = rowBase + k0 + sel * 8;
  v8h lo = *(const v8h*)(p);
  v8h hi = *(const v8h*)(p + 16);
  v16h r;
#pragma unroll
  for (int i = 0; i < 8; ++i) { r[i] = lo[i]; r[i + 8] = hi[i]; }
  return r;
}

__device__ __forceinline__ v8f wmma_f16(v16h a, v16h b, v8f c) {
  return __builtin_amdgcn_wmma_f32_16x16x32_f16(false, a, false, b, (short)0, c,
                                                false, false);
}

// Monotonic float -> uint key (descending order preserved by unsigned >=).
__device__ __forceinline__ unsigned fkey(float f) {
  unsigned b = __float_as_uint(f);
  return (b & 0x80000000u) ? ~b : (b | 0x80000000u);
}

// ---- f32 -> f16 convert ---------------------------------------------------
__global__ void cvt_f32_f16(const float* __restrict__ s, _Float16* __restrict__ d, int n) {
  int i = blockIdx.x * blockDim.x + threadIdx.x;
  if (i < n) d[i] = (_Float16)s[i];
}

// ---- NT GEMM: C[M,N] = alpha * A[M,K](f16) @ B[N,K]^T(f16), f32 accum -----
// One wave per 16(M) x 64(N) strip; A frag reused across 4 WMMAs; K-loop is
// register double-buffered so next loads are in flight during current WMMAs.
__global__ void wmma_gemm_nt(const _Float16* __restrict__ A,
                             const _Float16* __restrict__ B,
                             void* __restrict__ Cout,
                             int M, int N, int K, float alpha, int writeHalf) {
  const int gw   = (blockIdx.x * blockDim.x + threadIdx.x) >> 5;
  const int lane = threadIdx.x & 31;
  const int nStrips = N >> 6;
  const int total   = (M >> 4) * nStrips;
  if (gw >= total) return;
  const int m0  = (gw / nStrips) << 4;
  const int n0  = (gw % nStrips) << 6;
  const int sel = lane >> 4, l15 = lane & 15;

  const _Float16* aRow = A + (size_t)(m0 + l15) * K;
  const _Float16* bRow[4];
#pragma unroll
  for (int t = 0; t < 4; ++t) bRow[t] = B + (size_t)(n0 + t * 16 + l15) * K;

  // Warm L2 for this wave's strips (global_prefetch_b8).
  __builtin_prefetch(aRow, 0, 1);
#pragma unroll
  for (int t = 0; t < 4; ++t) __builtin_prefetch(bRow[t], 0, 1);

  v8f acc[4] = {};
  // Prologue: fragments for k0 = 0.
  v16h af = load_a_frag(aRow, sel, 0);
  v16h bf[4];
#pragma unroll
  for (int t = 0; t < 4; ++t) bf[t] = load_b_frag(bRow[t], sel, 0);

  for (int k0 = 0; k0 < K; k0 += 32) {
    // Branch-free prefetch of next K-step (clamped index on last iteration).
    const int kn = (k0 + 32 < K) ? (k0 + 32) : k0;
    v16h afn = load_a_frag(aRow, sel, kn);
    v16h bfn[4];
#pragma unroll
    for (int t = 0; t < 4; ++t) bfn[t] = load_b_frag(bRow[t], sel, kn);
    // Consume current fragments while next loads are in flight.
#pragma unroll
    for (int t = 0; t < 4; ++t) acc[t] = wmma_f16(af, bf[t], acc[t]);
    af = afn;
#pragma unroll
    for (int t = 0; t < 4; ++t) bf[t] = bfn[t];
  }

  // C/D layout: VGPR r -> row m0 + r + sel*8, col n0 + t*16 + (lane&15)
#pragma unroll
  for (int t = 0; t < 4; ++t) {
#pragma unroll
    for (int r = 0; r < 8; ++r) {
      const int row = m0 + r + sel * 8;
      const int col = n0 + t * 16 + l15;
      const float v = acc[t][r] * alpha;
      if (writeHalf) ((_Float16*)Cout)[(size_t)row * N + col] = (_Float16)v;
      else           ((float*)Cout)[(size_t)row * N + col]    = v;
    }
  }
}

// ---- Fused attention: WMMA logits -> LDS, radix-select top-32, softmax ----
// Grid: (L/16, H, B), 256 threads = 8 waves. Dyn LDS: 16*L*4 + lists.
__global__ void attn_topk_softmax(const _Float16* __restrict__ qp,   // [B*L, H*D] f16, pre-scaled
                                  const _Float16* __restrict__ kp,   // [B*L, H*D] f16
                                  const float*    __restrict__ vp,   // [B*L, H*D] f32
                                  float*          __restrict__ probs,// [B,H,L,L] f32
                                  _Float16*       __restrict__ mixed)// [B*L, H*D] f16
{
  constexpr int L = 2048, H = 16, D = 64, HD = 1024, TOPK = 32;
  extern __shared__ char smem[];
  float* logits  = (float*)smem;                                // [16][L]
  int*   listIdx = (int*)  (smem + (size_t)16 * L * 4);         // [8][64]
  float* listP   = (float*)(smem + (size_t)16 * L * 4 + 8 * 64 * 4);
  int*   cnts    = (int*)  (smem + (size_t)16 * L * 4 + 2 * 8 * 64 * 4);

  const int qt = blockIdx.x, h = blockIdx.y, b = blockIdx.z;
  const int wid = threadIdx.x >> 5, lane = threadIdx.x & 31;
  const int sel = lane >> 4, l15 = lane & 15;
  const int q0 = qt * 16;

  // ---- Phase 1: logits[16][2048] via WMMA (contraction over D=64 = 2 frags)
  const _Float16* qRow = qp + (size_t)(b * L + q0 + l15) * HD + h * D;
  __builtin_prefetch(qRow, 0, 1);
  v16h a0 = load_a_frag(qRow, sel, 0);
  v16h a1 = load_a_frag(qRow, sel, 32);

  // Register double-buffered sweep over 128 K-tiles (8 waves round-robin).
  {
    const _Float16* kRow = kp + (size_t)(b * L + wid * 16 + l15) * HD + h * D;
    v16h b0 = load_b_frag(kRow, sel, 0);
    v16h b1 = load_b_frag(kRow, sel, 32);
    for (int kt = wid; kt < L / 16; kt += 8) {
      const int ktn = (kt + 8 < L / 16) ? (kt + 8) : kt;   // branch-free clamp
      const _Float16* kRowN = kp + (size_t)(b * L + ktn * 16 + l15) * HD + h * D;
      v16h b0n = load_b_frag(kRowN, sel, 0);
      v16h b1n = load_b_frag(kRowN, sel, 32);
      v8f c = {};
      c = wmma_f16(a0, b0, c);
      c = wmma_f16(a1, b1, c);
      const int kb = kt * 16;
#pragma unroll
      for (int r = 0; r < 8; ++r)
        logits[(size_t)(r + sel * 8) * L + kb + l15] = c[r];
      b0 = b0n; b1 = b1n;
    }
  }
  __syncthreads();

  // ---- Phase 2: per-row top-32 + softmax + sparse probs@V (wave per row)
  int*   cnt  = cnts + wid;
  int*   lIdx = listIdx + wid * 64;
  float* lP   = listP + wid * 64;

  for (int row = wid; row < 16; row += 8) {
    if (lane == 0) *cnt = 0;
    // cache the row: 64 f32 per lane (strided by 32 for coalesced layout)
    float x[64]; unsigned u[64];
#pragma unroll
    for (int j = 0; j < 64; ++j) x[j] = logits[(size_t)row * L + lane + 32 * j];
    float m = x[0];
#pragma unroll
    for (int j = 1; j < 64; ++j) m = fmaxf(m, x[j]);
#pragma unroll
    for (int o = 16; o > 0; o >>= 1) m = fmaxf(m, __shfl_xor(m, o));
#pragma unroll
    for (int j = 0; j < 64; ++j) u[j] = fkey(x[j]);

    // radix binary search for the 32nd-largest key (pure VALU, no LDS traffic)
    unsigned prefix = 0;
    for (int bit = 31; bit >= 0; --bit) {
      const unsigned cand = prefix | (1u << bit);
      int c = 0;
#pragma unroll
      for (int j = 0; j < 64; ++j) c += (u[j] >= cand) ? 1 : 0;
#pragma unroll
      for (int o = 16; o > 0; o >>= 1) c += __shfl_xor(c, o);
      if (c >= TOPK) prefix = cand;
    }
    const unsigned T = prefix;

    // masked softmax (non-kept == exp(f32_min - max) == exactly 0)
    float s = 0.f;
#pragma unroll
    for (int j = 0; j < 64; ++j) {
      const float e = (u[j] >= T) ? __expf(x[j] - m) : 0.f;
      x[j] = e; s += e;
    }
#pragma unroll
    for (int o = 16; o > 0; o >>= 1) s += __shfl_xor(s, o);
    const float inv = 1.f / s;

    float* pr = probs + ((((size_t)b * H + h) * L) + q0 + row) * L;
    asm volatile("s_wait_dscnt 0x0" ::: "memory");  // *cnt reset visible
    __builtin_amdgcn_wave_barrier();
#pragma unroll 4
    for (int j = 0; j < 64; ++j) {
      const float p = x[j] * inv;
      pr[lane + 32 * j] = p;                        // coalesced HBM write (the big one)
      if (p > 0.f) {                                // ~32 survivors
        const int slot = atomicAdd(cnt, 1);
        if (slot < 64) { lIdx[slot] = lane + 32 * j; lP[slot] = p; }
      }
    }
    asm volatile("s_wait_dscnt 0x0" ::: "memory");  // list complete before reads
    __builtin_amdgcn_wave_barrier();

    int n = *cnt; if (n > 64) n = 64;
    float acc0 = 0.f, acc1 = 0.f;                   // lane owns d = lane, lane+32
    for (int i = 0; i < n; ++i) {
      const int   kc = lIdx[i];
      const float p  = lP[i];
      const float* vr = vp + (size_t)(b * L + kc) * HD + h * D;
      acc0 = fmaf(p, vr[lane],      acc0);
      acc1 = fmaf(p, vr[lane + 32], acc1);
    }
    _Float16* mr = mixed + (size_t)(b * L + q0 + row) * HD + h * D;
    mr[lane]      = (_Float16)acc0;
    mr[lane + 32] = (_Float16)acc1;
    asm volatile("s_wait_dscnt 0x0" ::: "memory");  // done with list before next row
    __builtin_amdgcn_wave_barrier();
  }
}

// ---------------------------------------------------------------------------
extern "C" void kernel_launch(void* const* d_in, const int* in_sizes, int n_in,
                              void* d_out, int out_size, void* d_ws, size_t ws_size,
                              hipStream_t stream) {
  (void)in_sizes; (void)n_in; (void)out_size; (void)ws_size;
  constexpr int Bb = 2, L = 2048, E = 1024, H = 16, D = 64, HD = 1024;
  constexpr int BL = Bb * L;  // 4096

  const float* q    = (const float*)d_in[0];
  const float* k    = (const float*)d_in[1];
  const float* v    = (const float*)d_in[2];
  const float* w_q  = (const float*)d_in[3];
  const float* w_k  = (const float*)d_in[4];
  const float* w_v  = (const float*)d_in[5];
  const float* w_fc = (const float*)d_in[6];

  char* ws = (char*)d_ws;
  size_t off = 0;
  auto alloc = [&](size_t bytes) {
    char* p = ws + off;
    off += (bytes + 255) & ~(size_t)255;
    return p;
  };
  _Float16* q16     = (_Float16*)alloc((size_t)BL * E * 2);
  _Float16* k16     = (_Float16*)alloc((size_t)BL * E * 2);
  _Float16* v16     = (_Float16*)alloc((size_t)BL * D * 2);
  _Float16* wq16    = (_Float16*)alloc((size_t)HD * E * 2);
  _Float16* wk16    = (_Float16*)alloc((size_t)HD * E * 2);
  _Float16* wv16    = (_Float16*)alloc((size_t)HD * D * 2);
  _Float16* wfc16   = (_Float16*)alloc((size_t)D * HD * 2);
  _Float16* qproj16 = (_Float16*)alloc((size_t)BL * HD * 2);
  _Float16* kproj16 = (_Float16*)alloc((size_t)BL * HD * 2);
  float*    vproj32 = (float*)   alloc((size_t)BL * HD * 4);
  _Float16* mixed16 = (_Float16*)alloc((size_t)BL * HD * 2);

  auto cvt = [&](const float* s, _Float16* d, int n) {
    cvt_f32_f16<<<(n + 255) / 256, 256, 0, stream>>>(s, d, n);
  };
  cvt(q,    q16,   BL * E);
  cvt(k,    k16,   BL * E);
  cvt(v,    v16,   BL * D);
  cvt(w_q,  wq16,  HD * E);
  cvt(w_k,  wk16,  HD * E);
  cvt(w_v,  wv16,  HD * D);
  cvt(w_fc, wfc16, D * HD);

  auto gemm = [&](const _Float16* A, const _Float16* Bm, void* C,
                  int M, int N, int K, float alpha, int wh) {
    const int waves  = (M >> 4) * (N >> 6);
    const int blocks = (waves * 32 + 255) / 256;
    wmma_gemm_nt<<<blocks, 256, 0, stream>>>(A, Bm, C, M, N, K, alpha, wh);
  };

  // Projections. Q pre-scaled by 1/(sqrt(D_K)*temperature) = 1/(8*0.7).
  gemm(q16, wq16, qproj16, BL, HD, E, 1.0f / (8.0f * 0.7f), 1);
  gemm(k16, wk16, kproj16, BL, HD, E, 1.0f, 1);
  gemm(v16, wv16, vproj32, BL, HD, D, 1.0f, 0);

  float* outp  = (float*)d_out;                 // [B*L, 64]
  float* probs = outp + (size_t)BL * D;         // [B,H,L,L]

  const size_t smemB = (size_t)16 * L * 4 + 2 * 8 * 64 * 4 + 8 * 4; // ~132KB
  attn_topk_softmax<<<dim3(L / 16, H, Bb), 256, smemB, stream>>>(
      qproj16, kproj16, vproj32, probs, mixed16);

  // Output projection: [BL, HD] @ w_fc[64, HD]^T -> [BL, 64] f32.
  gemm(mixed16, wfc16, outp, BL, D, HD, 1.0f, 0);
}
